// Attention_73358041416040
// MI455X (gfx1250) — matrix operand order repaired and checked by reading, not compiled, not run on hardware
//
#include <hip/hip_runtime.h>

#define DIMD 2048
#define SEQ  2048
#define BATCH 2
#define NH   16
#define HD   128

typedef __bf16 bf16_t;
typedef unsigned int u32;
typedef __attribute__((ext_vector_type(16))) __bf16 v16bf;
typedef __attribute__((ext_vector_type(8)))  __bf16 v8bf;
typedef __attribute__((ext_vector_type(8)))  float  v8f;
typedef __attribute__((ext_vector_type(4)))  u32    u32x4;
typedef __attribute__((ext_vector_type(4)))  int    i32x4;
typedef __attribute__((ext_vector_type(8)))  int    i32x8;

union V16 { v16bf v; v8bf h[2]; };

// ---------------- LDS raw offset (descriptor needs the LDS byte offset, not a
// generic aperture address) ----------------
__device__ inline u32 lds_offset_of(const void* p) {
  return (u32)(size_t)(__attribute__((address_space(3))) const char*)p;
}

// ---------------- Tensor Data Mover: 2D tile (tile_k x tile_rows) -> LDS ------
// D# per CDNA5 ISA: group0 = {count, lds_addr, global_addr,
// type=2}; group1 = {data_size=2B, tensor_dim0/1, tile_dim0/1, dim0_stride}.
__device__ inline void tdm_load_tile_2d(u32 lds_byte_off, const bf16_t* gptr,
                                        u32 tile_k, u32 tile_rows, u32 stride_elems) {
  unsigned long long ga = (unsigned long long)(size_t)gptr;
  u32x4 g0;
  g0[0] = 1u;                                            // count=1, user mode
  g0[1] = lds_byte_off;                                  // lds_addr (bytes)
  g0[2] = (u32)ga;                                       // global_addr[31:0]
  g0[3] = (u32)((ga >> 32) & 0x01ffffffu) | (2u << 30);  // addr[56:32] | type=2
  const u32 td0 = stride_elems;   // tensor extent dim0 (row length, elements)
  const u32 td1 = 1u << 20;       // rows: tiles are fully in-bounds
  i32x8 g1;
  g1[0] = (int)(1u << 16);                               // data_size=1 (2 bytes)
  g1[1] = (int)((td0 & 0xffffu) << 16);                  // tensor_dim0 lo16
  g1[2] = (int)((td0 >> 16) | ((td1 & 0xffffu) << 16));  // td0 hi16 | td1 lo16
  g1[3] = (int)((td1 >> 16) | (tile_k << 16));           // td1 hi16 | tile_dim0
  g1[4] = (int)tile_rows;                                // tile_dim1 (tile_dim2=0)
  g1[5] = (int)stride_elems;                             // tensor_dim0_stride lo32
  g1[6] = 0;
  g1[7] = 0;
  i32x4 z4 = {0, 0, 0, 0};
#if __has_include(<hip/amd_detail/amd_gfx1250_TDM.h>)
  i32x8 z8 = {0, 0, 0, 0, 0, 0, 0, 0};
  __builtin_amdgcn_tensor_load_to_lds(g0, g1, z4, z4, z8, 0);   // clang-23 arity
#else
  __builtin_amdgcn_tensor_load_to_lds(g0, g1, z4, z4, 0);       // ROCm 7.2 arity
#endif
}

// ---------------- fp32 -> bf16 conversion ----------------
__global__ void cvt_f32_bf16(const float* __restrict__ in, bf16_t* __restrict__ out, int n) {
  for (int i = blockIdx.x * blockDim.x + threadIdx.x; i < n; i += gridDim.x * blockDim.x)
    out[i] = (bf16_t)in[i];
}

// ---------------- NT GEMM: C[M,N] = A[M,K] * W[N,K]^T -------------------------
// Block = 8 waves -> 128x128 tile; wave -> 32x64. K staged in 64-wide chunks via
// TDM into double-buffered LDS (2 x (16KB A + 16KB B) = 64 KB of the 320 KB WGP).
template <bool OUT_BF16>
__global__ __launch_bounds__(256) void gemm_nt(const bf16_t* __restrict__ A,
                                               const bf16_t* __restrict__ W,
                                               void* __restrict__ Cout,
                                               int M, int N, int K) {
  constexpr int KC = 64;
  __shared__ alignas(16) bf16_t Alds[2][128 * KC];
  __shared__ alignas(16) bf16_t Blds[2][128 * KC];

  const int lane = threadIdx.x & 31;
  const int wave = threadIdx.x >> 5;
  const int hi = lane >> 4, ln = lane & 15;
  const int mSub = wave & 3;   // 32-row slice within block tile
  const int nSub = wave >> 2;  // 64-col slice within block tile
  const int mBase = blockIdx.y * 128;
  const int nBase = blockIdx.x * 128;
  const int nChunks = K / KC;

  v8f acc[2][4] = {};

  // prologue: stage chunk 0 into buffer 0 (TDM, EXEC-independent, per-wave issue)
  if (wave == 0) {
    tdm_load_tile_2d(lds_offset_of(&Alds[0][0]), A + (size_t)mBase * K, KC, 128, K);
    __builtin_amdgcn_s_wait_tensorcnt(0);
  } else if (wave == 1) {
    tdm_load_tile_2d(lds_offset_of(&Blds[0][0]), W + (size_t)nBase * K, KC, 128, K);
    __builtin_amdgcn_s_wait_tensorcnt(0);
  }
  __syncthreads();

  for (int c = 0; c < nChunks; ++c) {
    const int buf = c & 1;
    // overlap: kick off next chunk's TDM into the other buffer
    if (c + 1 < nChunks) {
      const int k0n = (c + 1) * KC;
      if (wave == 0)
        tdm_load_tile_2d(lds_offset_of(&Alds[buf ^ 1][0]), A + (size_t)mBase * K + k0n, KC, 128, K);
      else if (wave == 1)
        tdm_load_tile_2d(lds_offset_of(&Blds[buf ^ 1][0]), W + (size_t)nBase * K + k0n, KC, 128, K);
    }
    // compute on current buffer: 16 WMMA per wave per chunk
#pragma unroll
    for (int kk = 0; kk < KC; kk += 32) {
      V16 a[2];
#pragma unroll
      for (int mt = 0; mt < 2; ++mt) {
        const bf16_t* ap = &Alds[buf][(mSub * 32 + mt * 16 + ln) * KC + kk + hi * 8];
        a[mt].h[0] = *(const v8bf*)(ap);
        a[mt].h[1] = *(const v8bf*)(ap + 16);
      }
#pragma unroll
      for (int nt = 0; nt < 4; ++nt) {
        const bf16_t* bp = &Blds[buf][(nSub * 64 + nt * 16 + ln) * KC + kk + hi * 8];
        V16 b;
        b.h[0] = *(const v8bf*)(bp);
        b.h[1] = *(const v8bf*)(bp + 16);
        acc[0][nt] = __builtin_amdgcn_wmma_f32_16x16x32_bf16(false, a[0].v, false, b.v,
                                                             (short)0, acc[0][nt], false, false);
        acc[1][nt] = __builtin_amdgcn_wmma_f32_16x16x32_bf16(false, a[1].v, false, b.v,
                                                             (short)0, acc[1][nt], false, false);
      }
    }
    if (wave <= 1) __builtin_amdgcn_s_wait_tensorcnt(0);
    __syncthreads();
  }

  // epilogue: C layout VGPR i -> row (i + hi*8), lane -> col
#pragma unroll
  for (int mt = 0; mt < 2; ++mt) {
#pragma unroll
    for (int nt = 0; nt < 4; ++nt) {
#pragma unroll
      for (int i = 0; i < 8; ++i) {
        const size_t r = (size_t)(mBase + mSub * 32 + mt * 16 + i + hi * 8);
        const size_t cc = (size_t)(nBase + nSub * 64 + nt * 16 + ln);
        if (OUT_BF16) ((bf16_t*)Cout)[r * N + cc] = (bf16_t)acc[mt][nt][i];
        else          ((float*)Cout)[r * N + cc]  = acc[mt][nt][i];
      }
    }
  }
}

// ---------------- RoPE on bf16 Q and K ----------------
__global__ void rope_qk(bf16_t* __restrict__ Q, bf16_t* __restrict__ Kt,
                        const float* __restrict__ cosb, const float* __restrict__ sinb) {
  int idx = blockIdx.x * blockDim.x + threadIdx.x;   // over B*S*H*64 pairs
  const int i = idx & 63;
  const int h = (idx >> 6) & 15;
  const int s = (idx >> 10) & 2047;
  const int b = idx >> 21;
  const float c  = cosb[s * 64 + i];
  const float sn = sinb[s * 64 + i];
  const size_t off = (size_t)(b * SEQ + s) * DIMD + h * HD + 2 * i;
  {
    float te = (float)Q[off], to = (float)Q[off + 1];
    Q[off]     = (bf16_t)(te * c - to * sn);
    Q[off + 1] = (bf16_t)(te * sn + to * c);
  }
  {
    float te = (float)Kt[off], to = (float)Kt[off + 1];
    Kt[off]     = (bf16_t)(te * c - to * sn);
    Kt[off + 1] = (bf16_t)(te * sn + to * c);
  }
}

// ---------------- Flash attention (bf16 QKV, f32 online softmax) ----------------
__global__ __launch_bounds__(256) void flash_attn(const bf16_t* __restrict__ Q,
                                                  const bf16_t* __restrict__ Kt,
                                                  const bf16_t* __restrict__ V,
                                                  bf16_t* __restrict__ Oo) {
  __shared__ alignas(16) bf16_t Klds[32 * 128];     // [kv][d]
  __shared__ alignas(16) bf16_t VldsT[128 * 32];    // [d][kv]  (transposed for B-frags)
  __shared__ alignas(16) bf16_t Plds[8 * 16 * 32];  // per-wave P staging

  const float SCALE = 0.08838834764831845f;         // 1/sqrt(128)
  const int lane = threadIdx.x & 31;
  const int wave = threadIdx.x >> 5;
  const int hi   = lane >> 4;
  const int ln   = lane & 15;
  const int b = blockIdx.z, h = blockIdx.y;
  const int qTile = blockIdx.x * 8 + wave;

  V16 qa[4];
  {
    const bf16_t* qrow = Q + (size_t)(b * SEQ + qTile * 16 + ln) * DIMD + h * HD;
#pragma unroll
    for (int c = 0; c < 4; ++c) {
      const int ka = c * 32 + hi * 8;
      qa[c].h[0] = *(const v8bf*)(qrow + ka);
      qa[c].h[1] = *(const v8bf*)(qrow + ka + 16);
    }
  }

  float mrow[8], lrow[8];
#pragma unroll
  for (int i = 0; i < 8; ++i) { mrow[i] = -3.0e38f; lrow[i] = 0.f; }
  v8f o[8] = {};

  const int ldRow = threadIdx.x >> 3;
  const int ldCol = (threadIdx.x & 7) * 16;
  const bf16_t* Kg = Kt + (size_t)(b * SEQ) * DIMD + h * HD;
  const bf16_t* Vg = V  + (size_t)(b * SEQ) * DIMD + h * HD;

  for (int kv0 = 0; kv0 < SEQ; kv0 += 32) {
    {
      const bf16_t* kg = Kg + (size_t)(kv0 + ldRow) * DIMD + ldCol;
      *(v8bf*)(&Klds[ldRow * 128 + ldCol])     = *(const v8bf*)(kg);
      *(v8bf*)(&Klds[ldRow * 128 + ldCol + 8]) = *(const v8bf*)(kg + 8);
      const bf16_t* vg = Vg + (size_t)(kv0 + ldRow) * DIMD + ldCol;
      v8bf v0 = *(const v8bf*)(vg);
      v8bf v1 = *(const v8bf*)(vg + 8);
#pragma unroll
      for (int j = 0; j < 8; ++j) {
        VldsT[(ldCol + j) * 32 + ldRow]     = v0[j];
        VldsT[(ldCol + 8 + j) * 32 + ldRow] = v1[j];
      }
    }
    __syncthreads();

    v8f s0 = {}, s1 = {};
#pragma unroll
    for (int c = 0; c < 4; ++c) {
      V16 b0, b1;
      const bf16_t* kb0 = &Klds[ln * 128 + c * 32 + hi * 8];
      const bf16_t* kb1 = &Klds[(ln + 16) * 128 + c * 32 + hi * 8];
      b0.h[0] = *(const v8bf*)(kb0); b0.h[1] = *(const v8bf*)(kb0 + 16);
      b1.h[0] = *(const v8bf*)(kb1); b1.h[1] = *(const v8bf*)(kb1 + 16);
      s0 = __builtin_amdgcn_wmma_f32_16x16x32_bf16(false, qa[c].v, false, b0.v, (short)0, s0, false, false);
      s1 = __builtin_amdgcn_wmma_f32_16x16x32_bf16(false, qa[c].v, false, b1.v, (short)0, s1, false, false);
    }

    float mloc[8], p0[8], p1[8], rs[8], alpha[8];
#pragma unroll
    for (int i = 0; i < 8; ++i) {
      s0[i] *= SCALE; s1[i] *= SCALE;
      mloc[i] = fmaxf(s0[i], s1[i]);
    }
#pragma unroll
    for (int mask = 1; mask <= 8; mask <<= 1)
#pragma unroll
      for (int i = 0; i < 8; ++i)
        mloc[i] = fmaxf(mloc[i], __shfl_xor(mloc[i], mask, 32));
#pragma unroll
    for (int i = 0; i < 8; ++i) {
      float mnew = fmaxf(mrow[i], mloc[i]);
      alpha[i] = __expf(mrow[i] - mnew);
      p0[i] = __expf(s0[i] - mnew);
      p1[i] = __expf(s1[i] - mnew);
      rs[i] = p0[i] + p1[i];
      mrow[i] = mnew;
    }
#pragma unroll
    for (int mask = 1; mask <= 8; mask <<= 1)
#pragma unroll
      for (int i = 0; i < 8; ++i)
        rs[i] += __shfl_xor(rs[i], mask, 32);
#pragma unroll
    for (int i = 0; i < 8; ++i) lrow[i] = lrow[i] * alpha[i] + rs[i];
#pragma unroll
    for (int dt = 0; dt < 8; ++dt)
#pragma unroll
      for (int i = 0; i < 8; ++i) o[dt][i] *= alpha[i];

    bf16_t* pbase = &Plds[wave * 16 * 32];
#pragma unroll
    for (int i = 0; i < 8; ++i) {
      const int r = i + hi * 8;
      pbase[r * 32 + ln]      = (bf16_t)p0[i];
      pbase[r * 32 + 16 + ln] = (bf16_t)p1[i];
    }
    V16 pa;
    {
      const bf16_t* pr = pbase + ln * 32 + hi * 8;
      pa.h[0] = *(const v8bf*)(pr);
      pa.h[1] = *(const v8bf*)(pr + 16);
    }
#pragma unroll
    for (int dt = 0; dt < 8; ++dt) {
      V16 vb;
      const bf16_t* vp = &VldsT[(dt * 16 + ln) * 32 + hi * 8];
      vb.h[0] = *(const v8bf*)(vp);
      vb.h[1] = *(const v8bf*)(vp + 16);
      o[dt] = __builtin_amdgcn_wmma_f32_16x16x32_bf16(false, pa.v, false, vb.v, (short)0, o[dt], false, false);
    }
    __syncthreads();
  }

#pragma unroll
  for (int i = 0; i < 8; ++i) {
    const float inv = 1.0f / lrow[i];
    const size_t r = (size_t)(b * SEQ + qTile * 16 + i + hi * 8);
#pragma unroll
    for (int dt = 0; dt < 8; ++dt)
      Oo[r * DIMD + h * HD + dt * 16 + ln] = (bf16_t)(o[dt][i] * inv);
  }
}

// ---------------- launcher ----------------
extern "C" void kernel_launch(void* const* d_in, const int* in_sizes, int n_in,
                              void* d_out, int out_size, void* d_ws, size_t ws_size,
                              hipStream_t stream) {
  const float* x  = (const float*)d_in[0];
  const float* wq = (const float*)d_in[1];
  const float* wk = (const float*)d_in[2];
  const float* wv = (const float*)d_in[3];
  const float* wo = (const float*)d_in[4];
  const float* pc = (const float*)d_in[5];
  const float* ps = (const float*)d_in[6];

  const int M  = BATCH * SEQ;                 // 4096
  const size_t DD = (size_t)DIMD * DIMD;
  const size_t MD = (size_t)M * DIMD;

  char* p = (char*)d_ws;
  bf16_t* xb  = (bf16_t*)p; p += MD * 2;
  bf16_t* wqb = (bf16_t*)p; p += DD * 2;
  bf16_t* wkb = (bf16_t*)p; p += DD * 2;
  bf16_t* wvb = (bf16_t*)p; p += DD * 2;
  bf16_t* wob = (bf16_t*)p; p += DD * 2;
  bf16_t* Qb  = (bf16_t*)p; p += MD * 2;
  bf16_t* Kb  = (bf16_t*)p; p += MD * 2;
  bf16_t* Vb  = (bf16_t*)p; p += MD * 2;
  bf16_t* Ab  = (bf16_t*)p; p += MD * 2;

  cvt_f32_bf16<<<4096, 256, 0, stream>>>(x,  xb,  (int)MD);
  cvt_f32_bf16<<<4096, 256, 0, stream>>>(wq, wqb, (int)DD);
  cvt_f32_bf16<<<4096, 256, 0, stream>>>(wk, wkb, (int)DD);
  cvt_f32_bf16<<<4096, 256, 0, stream>>>(wv, wvb, (int)DD);
  cvt_f32_bf16<<<4096, 256, 0, stream>>>(wo, wob, (int)DD);

  dim3 gg(DIMD / 128, M / 128);
  gemm_nt<true><<<gg, 256, 0, stream>>>(xb, wqb, Qb, M, DIMD, DIMD);
  gemm_nt<true><<<gg, 256, 0, stream>>>(xb, wkb, Kb, M, DIMD, DIMD);
  gemm_nt<true><<<gg, 256, 0, stream>>>(xb, wvb, Vb, M, DIMD, DIMD);

  rope_qk<<<(BATCH * SEQ * NH * 64) / 256, 256, 0, stream>>>(Qb, Kb, pc, ps);

  dim3 ga(SEQ / 128, NH, BATCH);
  flash_attn<<<ga, 256, 0, stream>>>(Qb, Kb, Vb, Ab);

  gemm_nt<false><<<gg, 256, 0, stream>>>(Ab, wob, d_out, M, DIMD, DIMD);
}
